// DenseRNN_59957743452421
// MI455X (gfx1250) — matrix operand order repaired and checked by reading, not compiled, not run on hardware
//
#include <hip/hip_runtime.h>
#include <hip/hip_bf16.h>

// ---------------------------------------------------------------------------
// DenseRNN on MI455X (gfx1250, wave32, WMMA).
//   h_{t+1} = tanh([x_t, h_t] @ W_fc^T + b_fc)   (T=512 sequential steps)
//   y       = h_T @ W_out^T + b_out
// bf16 operands + v_wmma_f32_16x16x32_bf16 (fp32 accumulate).
// W_fc (3 MB bf16) stays L2-resident; unique HBM traffic ~1 GB => compute /
// latency bound, so all matmul FLOPs go through the WMMA pipe. The scan's
// cross-step dependency is realized as 512 stream-ordered launches (graph-
// capture safe; cluster barriers can't sync a whole grid).
// If ws_size permits (~139.5 MB) we pre-convert x to bf16 ONCE, removing the
// 64x-redundant per-step fp32->bf16 A-fragment conversions; otherwise we
// convert in-register per step (v_cvt_pk_bf16_f32 co-executes with WMMA).
// ---------------------------------------------------------------------------

typedef __attribute__((ext_vector_type(16))) __bf16 v16bf;
typedef __attribute__((ext_vector_type(8)))  __bf16 v8bf;
typedef __attribute__((ext_vector_type(8)))  float  v8f;

static constexpr int B_   = 256;
static constexpr int T_   = 512;
static constexpr int DIN  = 512;
static constexpr int DH   = 1024;
static constexpr int DOUT = 512;
static constexpr int KFC  = DIN + DH;   // 1536

// --- fragment loaders (ISA 7.12.2 VGPR layouts, wave32) --------------------

// A-fragment 16x32 bf16 from an fp32 row (lane holds row m = lane&15).
// elements 0..7  <-> K = kA..kA+7   (kA = k0 + (lane>>4)*8)
// elements 8..15 <-> K = kA+16..kA+23
__device__ inline v16bf a_frag_from_f32(const float* __restrict__ p, int kA) {
  const float4 f0 = *reinterpret_cast<const float4*>(p + kA);
  const float4 f1 = *reinterpret_cast<const float4*>(p + kA + 4);
  const float4 f2 = *reinterpret_cast<const float4*>(p + kA + 16);
  const float4 f3 = *reinterpret_cast<const float4*>(p + kA + 20);
  v16bf a;
  a[0]  = (__bf16)f0.x; a[1]  = (__bf16)f0.y; a[2]  = (__bf16)f0.z; a[3]  = (__bf16)f0.w;
  a[4]  = (__bf16)f1.x; a[5]  = (__bf16)f1.y; a[6]  = (__bf16)f1.z; a[7]  = (__bf16)f1.w;
  a[8]  = (__bf16)f2.x; a[9]  = (__bf16)f2.y; a[10] = (__bf16)f2.z; a[11] = (__bf16)f2.w;
  a[12] = (__bf16)f3.x; a[13] = (__bf16)f3.y; a[14] = (__bf16)f3.z; a[15] = (__bf16)f3.w;
  return a;
}

// Combine two contiguous 8x bf16 (16 B) loads into one 16-element fragment.
__device__ inline v16bf frag16(const __bf16* __restrict__ p0,
                               const __bf16* __restrict__ p1) {
  v8bf lo = *reinterpret_cast<const v8bf*>(p0);
  v8bf hi = *reinterpret_cast<const v8bf*>(p1);
  return __builtin_shufflevector(lo, hi, 0,1,2,3,4,5,6,7,8,9,10,11,12,13,14,15);
}

// --- fp32 -> bf16 conversion (weights / h0 / optional x prep) --------------
__global__ void cvt_f32_to_bf16(const float* __restrict__ src,
                                __bf16* __restrict__ dst, int n) {
  int i = blockIdx.x * blockDim.x + threadIdx.x;
  if (i < n) dst[i] = (__bf16)src[i];
}

// --- one recurrence step ---------------------------------------------------
// grid (8,16), block 256 (8 waves). wave -> one 16x16 tile of h_new.
// C[m,n] = tanh( sum_k A[m,k] * W_fc[n,k] + b_fc[n] ),
// A = concat(x[:,t,:], h_t): two K regions. XBF16 selects pre-converted x.
template <bool XBF16>
__global__ void rnn_step(const float* __restrict__ x_f32,   // [B,T,DIN] f32
                         const __bf16* __restrict__ x_bf,   // [B,T,DIN] bf16 (or null)
                         const __bf16* __restrict__ h_cur,  // [B,DH] bf16
                         const __bf16* __restrict__ Wfc,    // [DH,KFC] bf16
                         const float* __restrict__ b_fc,    // [DH] f32
                         __bf16* __restrict__ h_nxt,        // [B,DH] bf16
                         float* __restrict__ h_f32,         // [B,DH] f32 (d_out slice)
                         int t) {
  const int lane = threadIdx.x & 31;
  const int wave = threadIdx.x >> 5;
  const int half = lane >> 4;     // 0: lanes 0-15, 1: lanes 16-31
  const int lrow = lane & 15;

  const int n0 = (blockIdx.x * 8 + wave) * 16;  // N tile (DH)
  const int m0 = blockIdx.y * 16;               // M tile (B)

  v8f acc = {};

  const __bf16* __restrict__ wrow = Wfc + (size_t)(n0 + lrow) * KFC;
  const size_t xoff = (size_t)(m0 + lrow) * (T_ * DIN) + (size_t)t * DIN;

  // K region 1: x_t, K = 0..511
  if (XBF16) {
    const __bf16* __restrict__ xrow = x_bf + xoff;
    #pragma unroll 4
    for (int k0 = 0; k0 < DIN; k0 += 32) {
      const int kA = k0 + half * 8;
      v16bf a = frag16(xrow + kA, xrow + kA + 16);
      const int kB = k0 + half * 8;
      v16bf b = frag16(wrow + kB, wrow + kB + 16);
      acc = __builtin_amdgcn_wmma_f32_16x16x32_bf16(false, a, false, b,
                                                    (short)0, acc, false, false);
    }
  } else {
    const float* __restrict__ xrow = x_f32 + xoff;
    #pragma unroll 4
    for (int k0 = 0; k0 < DIN; k0 += 32) {
      v16bf a = a_frag_from_f32(xrow, k0 + half * 8);
      const int kB = k0 + half * 8;
      v16bf b = frag16(wrow + kB, wrow + kB + 16);
      acc = __builtin_amdgcn_wmma_f32_16x16x32_bf16(false, a, false, b,
                                                    (short)0, acc, false, false);
    }
  }

  // K region 2: h_t (bf16), K = 512..1535
  const __bf16* __restrict__ hrow = h_cur + (size_t)(m0 + lrow) * DH;
  #pragma unroll 4
  for (int k0 = 0; k0 < DH; k0 += 32) {
    const int kA = k0 + half * 8;
    v16bf a = frag16(hrow + kA, hrow + kA + 16);
    const int kB = DIN + k0 + half * 8;
    v16bf b = frag16(wrow + kB, wrow + kB + 16);
    acc = __builtin_amdgcn_wmma_f32_16x16x32_bf16(false, a, false, b,
                                                  (short)0, acc, false, false);
  }

  const float bias = b_fc[n0 + lrow];
  #pragma unroll
  for (int v = 0; v < 8; ++v) {
    const int m = m0 + half * 8 + v;          // C/D layout: VGPR v, lane half
    const float hv = tanhf(acc[v] + bias);
    const size_t idx = (size_t)m * DH + (n0 + lrow);
    h_f32[idx] = hv;
    h_nxt[idx] = (__bf16)hv;
  }
}

// --- output projection: y = h_T @ W_out^T + b_out --------------------------
// grid (4,16), block 256. M=256, N=512, K=1024.
__global__ void rnn_out(const __bf16* __restrict__ h_fin,  // [B,DH] bf16
                        const __bf16* __restrict__ Wout,   // [DOUT,DH] bf16
                        const float* __restrict__ b_out,   // [DOUT]
                        float* __restrict__ y) {           // [B,DOUT]
  const int lane = threadIdx.x & 31;
  const int wave = threadIdx.x >> 5;
  const int half = lane >> 4;
  const int lrow = lane & 15;

  const int n0 = (blockIdx.x * 8 + wave) * 16;
  const int m0 = blockIdx.y * 16;

  v8f acc = {};
  const __bf16* __restrict__ wrow = Wout + (size_t)(n0 + lrow) * DH;
  const __bf16* __restrict__ hrow = h_fin + (size_t)(m0 + lrow) * DH;

  #pragma unroll 4
  for (int k0 = 0; k0 < DH; k0 += 32) {
    const int kA = k0 + half * 8;
    v16bf a = frag16(hrow + kA, hrow + kA + 16);
    const int kB = k0 + half * 8;
    v16bf b = frag16(wrow + kB, wrow + kB + 16);
    acc = __builtin_amdgcn_wmma_f32_16x16x32_bf16(false, a, false, b,
                                                  (short)0, acc, false, false);
  }

  const float bias = b_out[n0 + lrow];
  #pragma unroll
  for (int v = 0; v < 8; ++v) {
    const int m = m0 + half * 8 + v;
    y[(size_t)m * DOUT + (n0 + lrow)] = acc[v] + bias;
  }
}

// ---------------------------------------------------------------------------
extern "C" void kernel_launch(void* const* d_in, const int* in_sizes, int n_in,
                              void* d_out, int out_size, void* d_ws, size_t ws_size,
                              hipStream_t stream) {
  const float* x     = (const float*)d_in[0];  // [B,T,DIN]
  const float* h0    = (const float*)d_in[1];  // [B,DH]
  const float* W_fc  = (const float*)d_in[2];  // [DH,KFC]
  const float* b_fc  = (const float*)d_in[3];  // [DH]
  const float* W_out = (const float*)d_in[4];  // [DOUT,DH]
  const float* b_out = (const float*)d_in[5];  // [DOUT]

  float* y     = (float*)d_out;                // [B,DOUT]
  float* h_f32 = (float*)d_out + B_ * DOUT;    // [B,DH] (second tuple output)

  // Workspace layout (bytes):
  //   Wfc bf16 (3 MB) | Wout bf16 (1 MB) | h bf16 ping | h bf16 pong
  //   [optional] x bf16 (128 MB) if ws_size allows
  char* ws = (char*)d_ws;
  __bf16* Wfc_bf  = (__bf16*)(ws);
  __bf16* Wout_bf = (__bf16*)(ws + (size_t)DH * KFC * 2);
  __bf16* hb0     = (__bf16*)(ws + (size_t)DH * KFC * 2 + (size_t)DOUT * DH * 2);
  __bf16* hb1     = hb0 + (size_t)B_ * DH;
  __bf16* hb[2]   = { hb0, hb1 };
  const size_t base_bytes = (size_t)DH * KFC * 2 + (size_t)DOUT * DH * 2
                          + 2u * (size_t)B_ * DH * 2;      // 5,242,880
  const size_t x_elems = (size_t)B_ * T_ * DIN;            // 67,108,864
  const bool   x_in_ws = (ws_size >= base_bytes + x_elems * 2);
  __bf16* x_bf = (__bf16*)(ws + base_bytes);               // valid iff x_in_ws

  const int n_wfc = DH * KFC;       // 1,572,864
  const int n_wo  = DOUT * DH;      //   524,288
  const int n_h   = B_ * DH;        //   262,144
  cvt_f32_to_bf16<<<(n_wfc + 255) / 256, 256, 0, stream>>>(W_fc, Wfc_bf, n_wfc);
  cvt_f32_to_bf16<<<(n_wo  + 255) / 256, 256, 0, stream>>>(W_out, Wout_bf, n_wo);
  cvt_f32_to_bf16<<<(n_h   + 255) / 256, 256, 0, stream>>>(h0, hb[0], n_h);
  if (x_in_ws) {
    cvt_f32_to_bf16<<<(unsigned)((x_elems + 255) / 256), 256, 0, stream>>>(
        x, x_bf, (int)x_elems);
  }

  // Sequential scan: stream ordering provides the device-wide per-step sync.
  for (int t = 0; t < T_; ++t) {
    if (x_in_ws) {
      rnn_step<true><<<dim3(8, 16), 256, 0, stream>>>(
          x, x_bf, hb[t & 1], Wfc_bf, b_fc, hb[(t + 1) & 1], h_f32, t);
    } else {
      rnn_step<false><<<dim3(8, 16), 256, 0, stream>>>(
          x, nullptr, hb[t & 1], Wfc_bf, b_fc, hb[(t + 1) & 1], h_f32, t);
    }
  }
  // After T_=512 steps the final h lives in hb[T_ & 1] (fp32 copy in h_f32).
  rnn_out<<<dim3(4, 16), 256, 0, stream>>>(hb[T_ & 1], Wout_bf, b_out, y);
}
